// SwinTransformerBlock_8658654069043
// MI455X (gfx1250) — matrix operand order repaired and checked by reading, not compiled, not run on hardware
//
#include <hip/hip_runtime.h>
#include <hip/hip_bf16.h>

typedef __attribute__((ext_vector_type(16))) __bf16       v16bf;
typedef __attribute__((ext_vector_type(8)))  float        v8f;
typedef __attribute__((ext_vector_type(4)))  unsigned int u32x4;

union FragBF { u32x4 q[2]; v16bf v; __bf16 e[16]; };

__device__ __forceinline__ v8f v8f_zero() {
  v8f z = {0.f,0.f,0.f,0.f,0.f,0.f,0.f,0.f};
  return z;
}
__device__ __forceinline__ u32x4 u4_zero() {
  u32x4 z = {0u,0u,0u,0u};
  return z;
}

// ---------------------------------------------------------------------------
// CDNA5 async copy: global -> LDS, 16B per lane, tracked by ASYNCcnt.
// LDS destination operand is the wave-relative LDS byte offset; for flat
// shared pointers the low 32 bits are exactly that offset (LDS aperture
// mapping discards the upper bits).
// ---------------------------------------------------------------------------
__device__ __forceinline__ void async_copy_b128(__bf16* lds_dst, const __bf16* gsrc) {
  unsigned loff = (unsigned)(unsigned long long)(uintptr_t)lds_dst;
  unsigned long long ga = (unsigned long long)(uintptr_t)gsrc;
  asm volatile("global_load_async_to_lds_b128 %0, %1, off"
               :: "v"(loff), "v"(ga) : "memory");
}
__device__ __forceinline__ void wait_async0() {
  asm volatile("s_wait_asynccnt 0x0" ::: "memory");
}

// ---------------------------------------------------------------------------
// f32 -> bf16 weight conversion
// ---------------------------------------------------------------------------
__global__ __launch_bounds__(256) void cvt_bf16_kernel(
    const float* __restrict__ s, __bf16* __restrict__ d, int n) {
  int i = blockIdx.x * 256 + threadIdx.x;
  if (i < n) d[i] = (__bf16)s[i];
}

// ---------------------------------------------------------------------------
// LN1 fused with cyclic shift (-3,-3) + 7x7 window partition. One block per
// window-token row (widx*49+t); gathers the source pixel, LayerNorms over 384
// channels, writes bf16.
// ---------------------------------------------------------------------------
__global__ __launch_bounds__(128) void ln1_shift_kernel(
    const float* __restrict__ x, const float* __restrict__ g,
    const float* __restrict__ b, __bf16* __restrict__ out) {
  const int mt   = blockIdx.x;          // widx*49 + t
  const int widx = mt / 49;
  const int t    = mt - widx * 49;
  const int w    = widx & 63;
  const int bimg = widx >> 6;
  const int wh = w >> 3, ww = w & 7;
  const int tr = t / 7, tc = t - tr * 7;
  int r = wh * 7 + tr + 3; if (r >= 56) r -= 56;   // shifted[R] = orig[(R+3)%56]
  int c = ww * 7 + tc + 3; if (c >= 56) c -= 56;
  const size_t srow = (size_t)bimg * 3136 + (size_t)(r * 56 + c);
  const float* src = x + srow * 384;
  const int tid = threadIdx.x;
  float v0 = src[tid], v1 = src[tid + 128], v2 = src[tid + 256];
  __shared__ float rs[128], rq[128];
  rs[tid] = v0 + v1 + v2;
  rq[tid] = v0 * v0 + v1 * v1 + v2 * v2;
  __syncthreads();
  for (int o = 64; o > 0; o >>= 1) {
    if (tid < o) { rs[tid] += rs[tid + o]; rq[tid] += rq[tid + o]; }
    __syncthreads();
  }
  float mean = rs[0] * (1.f / 384.f);
  float var  = rq[0] * (1.f / 384.f) - mean * mean;
  float rstd = rsqrtf(var + 1e-5f);
  __bf16* dst = out + (size_t)mt * 384;
  dst[tid]       = (__bf16)((v0 - mean) * rstd * g[tid]       + b[tid]);
  dst[tid + 128] = (__bf16)((v1 - mean) * rstd * g[tid + 128] + b[tid + 128]);
  dst[tid + 256] = (__bf16)((v2 - mean) * rstd * g[tid + 256] + b[tid + 256]);
}

// ---------------------------------------------------------------------------
// LN2: plain row-major LayerNorm f32 -> bf16
// ---------------------------------------------------------------------------
__global__ __launch_bounds__(128) void ln_rows_kernel(
    const float* __restrict__ x, const float* __restrict__ g,
    const float* __restrict__ b, __bf16* __restrict__ out) {
  const size_t mt = blockIdx.x;
  const float* src = x + mt * 384;
  const int tid = threadIdx.x;
  float v0 = src[tid], v1 = src[tid + 128], v2 = src[tid + 256];
  __shared__ float rs[128], rq[128];
  rs[tid] = v0 + v1 + v2;
  rq[tid] = v0 * v0 + v1 * v1 + v2 * v2;
  __syncthreads();
  for (int o = 64; o > 0; o >>= 1) {
    if (tid < o) { rs[tid] += rs[tid + o]; rq[tid] += rq[tid + o]; }
    __syncthreads();
  }
  float mean = rs[0] * (1.f / 384.f);
  float var  = rq[0] * (1.f / 384.f) - mean * mean;
  float rstd = rsqrtf(var + 1e-5f);
  __bf16* dst = out + mt * 384;
  dst[tid]       = (__bf16)((v0 - mean) * rstd * g[tid]       + b[tid]);
  dst[tid + 128] = (__bf16)((v1 - mean) * rstd * g[tid + 128] + b[tid + 128]);
  dst[tid + 256] = (__bf16)((v2 - mean) * rstd * g[tid + 256] + b[tid + 256]);
}

// ---------------------------------------------------------------------------
// Tiled WMMA GEMM:  out[M,N] = A[M,K](bf16) @ W[N,K]^T(bf16) + bias
// Block tile 128x128, K-step 32, 8 waves each computing 64x32 via 4x2
// v_wmma_f32_16x16x32_bf16 fragments. Double-buffered LDS; the streaming
// A tile rides GLOBAL_LOAD_ASYNC_TO_LDS_B128 (ASYNCcnt) so the next tile's
// HBM fetch overlaps the current tile's WMMAs, with ONE barrier per K-step.
// W staged k-major (transposed, stride 144); B-operand fragments are 16
// strided bf16 reads per lane (lane = n, regs = k).
// Requires M % 128 == 0, N % 128 == 0, K % 32 == 0 (true for all launches).
// Epilogues: 0 = bf16 store (qkv)
//            1 = exact GELU -> bf16 (fc1)
//            2 = window-reverse + unshift scatter + residual -> f32 (proj)
//            3 = residual add -> f32 (fc2, final output)
// ---------------------------------------------------------------------------
template <int EPI>
__global__ __launch_bounds__(256) void gemm_bf16_kernel(
    const __bf16* __restrict__ A, const __bf16* __restrict__ W,
    const float* __restrict__ bias, int M, int N, int K,
    __bf16* __restrict__ obf, const float* __restrict__ resid,
    float* __restrict__ of) {
  __shared__ __bf16 lA[2][128 * 40];
  __shared__ __bf16 lB[2][32 * 144];
  const int tid = threadIdx.x;
  const int l = tid & 31, lo = l & 15, hi = l >> 4;
  const int wv  = tid >> 5;
  const int bm0 = blockIdx.y * 128, bn0 = blockIdx.x * 128;
  const int wm0 = (wv >> 2) * 64, wn0 = (wv & 3) * 32;
  const int srow = tid >> 1, skh = (tid & 1) * 16;

  v8f acc[4][2];
#pragma unroll
  for (int i = 0; i < 4; ++i)
#pragma unroll
    for (int j = 0; j < 2; ++j) acc[i][j] = v8f_zero();

  const __bf16* gpa = A + (size_t)(bm0 + srow) * K + skh;  // A stream (HBM-bound)
  const __bf16* gpb = W + (size_t)(bn0 + srow) * K + skh;  // W stream (L2-resident)

  const int nk = K >> 5;

  // ---- prologue: stage B tile 0 (sync, transposed) + launch async A tile 0
  {
    const u32x4* p = (const u32x4*)gpb;
    FragBF fb; fb.q[0] = p[0]; fb.q[1] = p[1];
#pragma unroll
    for (int i = 0; i < 16; ++i) lB[0][(skh + i) * 144 + srow] = fb.e[i];
    async_copy_b128(&lA[0][srow * 40 + skh], gpa);
    async_copy_b128(&lA[0][srow * 40 + skh + 8], gpa + 8);
  }

  for (int kt = 0; kt < nk; ++kt) {
    const int cur = kt & 1, nxt = cur ^ 1;
    const bool hasnext = (kt + 1) < nk;

    // next B tile -> registers (overlaps the wait below)
    u32x4 b0 = u4_zero(), b1 = b0;
    if (hasnext) {
      const u32x4* p = (const u32x4*)(gpb + (kt + 1) * 32);
      b0 = p[0]; b1 = p[1];
      __builtin_prefetch(gpb + (kt + 2) * 32, 0, 0);  // CDNA5 global_prefetch
    }

    wait_async0();      // this wave's A tile kt has landed in LDS
    __syncthreads();    // publish tile kt; everyone done reading buffer `nxt`

    if (hasnext) {
      // async A tile kt+1 into the other buffer (overlaps compute below)
      const __bf16* ga = gpa + (kt + 1) * 32;
      async_copy_b128(&lA[nxt][srow * 40 + skh], ga);
      async_copy_b128(&lA[nxt][srow * 40 + skh + 8], ga + 8);
      // transposed B stores into the other buffer
      FragBF fb; fb.q[0] = b0; fb.q[1] = b1;
#pragma unroll
      for (int i = 0; i < 16; ++i) lB[nxt][(skh + i) * 144 + srow] = fb.e[i];
    }

    // ---- compute tile kt from the current buffers
    v16bf af[4], bf2[2];
#pragma unroll
    for (int im = 0; im < 4; ++im) {
      const __bf16* p = &lA[cur][(wm0 + im * 16 + lo) * 40 + hi * 8];
      FragBF f; f.q[0] = *(const u32x4*)p; f.q[1] = *(const u32x4*)(p + 16);
      af[im] = f.v;
    }
#pragma unroll
    for (int jn = 0; jn < 2; ++jn) {
      FragBF f;
#pragma unroll
      for (int s = 0; s < 16; ++s) f.e[s] = lB[cur][((s < 8 ? 0 : 8) + hi * 8 + s) * 144 + wn0 + jn * 16 + lo];
      bf2[jn] = f.v;
    }
#pragma unroll
    for (int im = 0; im < 4; ++im)
#pragma unroll
      for (int jn = 0; jn < 2; ++jn)
        acc[im][jn] = __builtin_amdgcn_wmma_f32_16x16x32_bf16(
            false, af[im], false, bf2[jn], (short)0, acc[im][jn], false, false);
  }

#pragma unroll
  for (int im = 0; im < 4; ++im) {
#pragma unroll
    for (int r = 0; r < 8; ++r) {
      const int m = bm0 + wm0 + im * 16 + r + hi * 8;
      size_t drow = (size_t)m;
      if (EPI == 2) {  // window reverse + roll(+3,+3)
        const int widx = m / 49;
        const int t = m - widx * 49;
        const int w = widx & 63, bimg = widx >> 6;
        const int wh = w >> 3, ww = w & 7;
        const int tr = t / 7, tc = t - tr * 7;
        int rr = wh * 7 + tr + 3; if (rr >= 56) rr -= 56;
        int cc = ww * 7 + tc + 3; if (cc >= 56) cc -= 56;
        drow = (size_t)bimg * 3136 + (size_t)(rr * 56 + cc);
      }
#pragma unroll
      for (int jn = 0; jn < 2; ++jn) {
        const int n = bn0 + wn0 + jn * 16 + lo;
        float v = acc[im][jn][r] + bias[n];
        if (EPI == 0) {
          obf[(size_t)m * N + n] = (__bf16)v;
        } else if (EPI == 1) {
          float gel = 0.5f * v * (1.f + erff(v * 0.70710678118654752f));
          obf[(size_t)m * N + n] = (__bf16)gel;
        } else if (EPI == 2) {
          of[drow * 384 + n] = v + resid[drow * 384 + n];
        } else {
          of[(size_t)m * N + n] = v + resid[(size_t)m * N + n];
        }
      }
    }
  }
}

// ---------------------------------------------------------------------------
// Window attention. Grid (3, 2048): blockIdx.y = window, 4 waves x 3 = 12
// heads. Computes S^T = K(64x32) @ Q^T(32x64) so that softmax rows (keys)
// land in per-lane registers of the C/D layout; applies rel-pos bias and
// shift mask analytically; writes P (query-major) back to LDS as the
// A-operand of O = P @ V; V B-operand fragments load straight from global
// (lane -> d column, 16 strided key rows per lane). N=49 padded to 64.
// ---------------------------------------------------------------------------
__global__ __launch_bounds__(128) void attn_kernel(
    const __bf16* __restrict__ qkv, const float* __restrict__ rpb,
    __bf16* __restrict__ out) {
  __shared__ __bf16 sQt[4][32 * 72];
  __shared__ __bf16 sP [4][64 * 72];
  const int tid = threadIdx.x;
  const int l = tid & 31, lo = l & 15, hi = l >> 4;
  const int wv = tid >> 5;
  const int widx = blockIdx.y;
  const int h = blockIdx.x * 4 + wv;
  const int w = widx & 63;
  const int wh = w >> 3, ww = w & 7;
  __bf16* Qt = sQt[wv];
  __bf16* P  = sP[wv];

  const __bf16* qg = qkv + (size_t)widx * 49 * 1152 + h * 32;
  const __bf16* kg = qg + 384;
  const __bf16* vg = qg + 768;

  // stage scaled Q^T into LDS: Qt[d][m], stride 72
  for (int idx = l; idx < 64 * 32; idx += 32) {
    const int m = idx >> 5, d = idx & 31;
    float v = (m < 49) ? (float)qg[(size_t)m * 1152 + d] * 0.17677669529663687f : 0.f;
    Qt[d * 72 + m] = (__bf16)v;
  }

  // K rows as A operand (rows n, K-dim d)
  v16bf ka[4];
#pragma unroll
  for (int it = 0; it < 4; ++it) {
    const int n = it * 16 + lo;
    FragBF f;
    if (n < 49) {
      const __bf16* p = kg + (size_t)n * 1152 + hi * 8;
      f.q[0] = *(const u32x4*)p; f.q[1] = *(const u32x4*)(p + 16);
    } else { f.q[0] = u4_zero(); f.q[1] = u4_zero(); }
    ka[it] = f.v;
  }
  // Q^T as B operand (lane = query m, regs = d in table order)
  v16bf qb[4];
#pragma unroll
  for (int jm = 0; jm < 4; ++jm) {
    FragBF f;
#pragma unroll
    for (int s = 0; s < 16; ++s) f.e[s] = Qt[((s < 8 ? 0 : 8) + hi * 8 + s) * 72 + jm * 16 + lo];
    qb[jm] = f.v;
  }

  v8f t[4][4];
#pragma unroll
  for (int it = 0; it < 4; ++it)
#pragma unroll
    for (int jm = 0; jm < 4; ++jm) {
      t[it][jm] = v8f_zero();
      t[it][jm] = __builtin_amdgcn_wmma_f32_16x16x32_bf16(
          false, ka[it], false, qb[jm], (short)0, t[it][jm], false, false);
    }

  // T[n][m] == attn[m][n]: add rel-pos bias rp[h][m][n] and shift mask
#pragma unroll
  for (int it = 0; it < 4; ++it) {
#pragma unroll
    for (int r = 0; r < 8; ++r) {
      const int n = it * 16 + r + hi * 8;
      const int nr = n / 7, nc = n - nr * 7;
#pragma unroll
      for (int jm = 0; jm < 4; ++jm) {
        const int m = jm * 16 + lo;
        float v = t[it][jm][r];
        if (n < 49) {
          if (m < 49) {
            const int mr = m / 7, mc = m - mr * 7;
            const int idx = ((mr - nr + 6) * 13 + (mc - nc + 6)) * 12 + h;
            v += rpb[idx];
            const int Rm = wh * 7 + mr, Cm = ww * 7 + mc;
            const int Rn = wh * 7 + nr, Cn = ww * 7 + nc;
            const int gm_ = (Rm < 49 ? 0 : (Rm < 53 ? 1 : 2)) * 3 + (Cm < 49 ? 0 : (Cm < 53 ? 1 : 2));
            const int gn_ = (Rn < 49 ? 0 : (Rn < 53 ? 1 : 2)) * 3 + (Cn < 49 ? 0 : (Cn < 53 ? 1 : 2));
            if (gm_ != gn_) v -= 100.f;
          }
        } else {
          v = -1e30f;  // padded keys -> softmax weight 0
        }
        t[it][jm][r] = v;
      }
    }
  }

  // softmax over keys: per-lane over 32 regs + one xor-16 partner combine
  float inv[4];
#pragma unroll
  for (int jm = 0; jm < 4; ++jm) {
    float mx = -3.0e38f;
#pragma unroll
    for (int it = 0; it < 4; ++it)
#pragma unroll
      for (int r = 0; r < 8; ++r) mx = fmaxf(mx, t[it][jm][r]);
    mx = fmaxf(mx, __shfl_xor(mx, 16, 32));
    float s = 0.f;
#pragma unroll
    for (int it = 0; it < 4; ++it)
#pragma unroll
      for (int r = 0; r < 8; ++r) {
        float e = __expf(t[it][jm][r] - mx);
        t[it][jm][r] = e;
        s += e;
      }
    s += __shfl_xor(s, 16, 32);
    inv[jm] = 1.f / s;
  }

  // transpose-store P[m][n] (bf16) into LDS as A-operand source
#pragma unroll
  for (int it = 0; it < 4; ++it)
#pragma unroll
    for (int r = 0; r < 8; ++r) {
      const int n = it * 16 + r + hi * 8;
#pragma unroll
      for (int jm = 0; jm < 4; ++jm) {
        const int m = jm * 16 + lo;
        P[m * 72 + n] = (__bf16)(t[it][jm][r] * inv[jm]);
      }
    }

  // O = P(64x64) @ V(64x32)
  v8f o[4][2];
#pragma unroll
  for (int im = 0; im < 4; ++im)
#pragma unroll
    for (int jn = 0; jn < 2; ++jn) o[im][jn] = v8f_zero();

#pragma unroll
  for (int ks = 0; ks < 2; ++ks) {
    const int k0 = ks * 32;
    v16bf pa[4], vb[2];
#pragma unroll
    for (int im = 0; im < 4; ++im) {
      const __bf16* p = &P[(im * 16 + lo) * 72 + k0 + hi * 8];
      FragBF f; f.q[0] = *(const u32x4*)p; f.q[1] = *(const u32x4*)(p + 16);
      pa[im] = f.v;
    }
    // B-operand: lane = d column, regs = key row in table order
#pragma unroll
    for (int jn = 0; jn < 2; ++jn) {
      FragBF f;
#pragma unroll
      for (int s = 0; s < 16; ++s) {
        const int n = k0 + (s < 8 ? 0 : 8) + hi * 8 + s;
        f.e[s] = (n < 49) ? vg[(size_t)n * 1152 + jn * 16 + lo] : (__bf16)0.f;
      }
      vb[jn] = f.v;
    }
#pragma unroll
    for (int im = 0; im < 4; ++im)
#pragma unroll
      for (int jn = 0; jn < 2; ++jn)
        o[im][jn] = __builtin_amdgcn_wmma_f32_16x16x32_bf16(
            false, pa[im], false, vb[jn], (short)0, o[im][jn], false, false);
  }

#pragma unroll
  for (int im = 0; im < 4; ++im)
#pragma unroll
    for (int r = 0; r < 8; ++r) {
      const int m = im * 16 + r + hi * 8;
      if (m < 49) {
#pragma unroll
        for (int jn = 0; jn < 2; ++jn) {
          const int d = jn * 16 + lo;
          out[((size_t)widx * 49 + m) * 384 + h * 32 + d] = (__bf16)o[im][jn][r];
        }
      }
    }
}

// ---------------------------------------------------------------------------
extern "C" void kernel_launch(void* const* d_in, const int* in_sizes, int n_in,
                              void* d_out, int out_size, void* d_ws, size_t ws_size,
                              hipStream_t stream) {
  (void)in_sizes; (void)n_in; (void)out_size; (void)ws_size;
  const float* x      = (const float*)d_in[0];
  const float* n1g    = (const float*)d_in[1];
  const float* n1b    = (const float*)d_in[2];
  const float* qkv_w  = (const float*)d_in[3];
  const float* qkv_b  = (const float*)d_in[4];
  const float* proj_w = (const float*)d_in[5];
  const float* proj_b = (const float*)d_in[6];
  const float* rpb    = (const float*)d_in[7];
  const float* n2g    = (const float*)d_in[8];
  const float* n2b    = (const float*)d_in[9];
  const float* fc1_w  = (const float*)d_in[10];
  const float* fc1_b  = (const float*)d_in[11];
  const float* fc2_w  = (const float*)d_in[12];
  const float* fc2_b  = (const float*)d_in[13];

  char* ws = (char*)d_ws;
  const size_t M = 100352;  // 32 * 56 * 56  (== 2048 windows * 49 tokens)
  size_t off = 0;
  __bf16* w_qkv  = (__bf16*)(ws + off); off += (size_t)1152 * 384 * 2;
  __bf16* w_proj = (__bf16*)(ws + off); off += (size_t)384 * 384 * 2;
  __bf16* w_fc1  = (__bf16*)(ws + off); off += (size_t)1536 * 384 * 2;
  __bf16* w_fc2  = (__bf16*)(ws + off); off += (size_t)384 * 1536 * 2;
  __bf16* bufA   = (__bf16*)(ws + off); off += M * 384 * 2;    // xw / attn-out / ln2-out
  __bf16* bufB   = (__bf16*)(ws + off); off += M * 1536 * 2;   // qkv / fc1-out
  float*  x1     = (float*)(ws + off);  off += M * 384 * 4;    // attn residual stream

  // weights -> bf16
  cvt_bf16_kernel<<<(1152 * 384 + 255) / 256, 256, 0, stream>>>(qkv_w, w_qkv, 1152 * 384);
  cvt_bf16_kernel<<<(384 * 384 + 255) / 256, 256, 0, stream>>>(proj_w, w_proj, 384 * 384);
  cvt_bf16_kernel<<<(1536 * 384 + 255) / 256, 256, 0, stream>>>(fc1_w, w_fc1, 1536 * 384);
  cvt_bf16_kernel<<<(384 * 1536 + 255) / 256, 256, 0, stream>>>(fc2_w, w_fc2, 384 * 1536);

  // LN1 + shift + window partition -> bf16
  ln1_shift_kernel<<<(unsigned)M, 128, 0, stream>>>(x, n1g, n1b, bufA);
  // qkv GEMM: (100352 x 384) @ (1152 x 384)^T
  gemm_bf16_kernel<0><<<dim3(1152 / 128, (unsigned)(M / 128)), 256, 0, stream>>>(
      bufA, w_qkv, qkv_b, (int)M, 1152, 384, bufB, nullptr, nullptr);
  // window attention (2048 windows x 12 heads)
  attn_kernel<<<dim3(3, 2048), 128, 0, stream>>>(bufB, rpb, bufA);
  // proj GEMM + window-reverse + unshift + residual -> x1 (f32)
  gemm_bf16_kernel<2><<<dim3(384 / 128, (unsigned)(M / 128)), 256, 0, stream>>>(
      bufA, w_proj, proj_b, (int)M, 384, 384, nullptr, x, x1);
  // LN2 -> bf16
  ln_rows_kernel<<<(unsigned)M, 128, 0, stream>>>(x1, n2g, n2b, bufA);
  // fc1 GEMM + exact GELU -> bf16
  gemm_bf16_kernel<1><<<dim3(1536 / 128, (unsigned)(M / 128)), 256, 0, stream>>>(
      bufA, w_fc1, fc1_b, (int)M, 1536, 384, bufB, nullptr, nullptr);
  // fc2 GEMM + residual -> f32 output
  gemm_bf16_kernel<3><<<dim3(384 / 128, (unsigned)(M / 128)), 256, 0, stream>>>(
      bufB, w_fc2, fc2_b, (int)M, 384, 1536, nullptr, x1, (float*)d_out);
}
